// LoFTREncoderLayer_63359357551218
// MI455X (gfx1250) — compile-verified, matrix operand order checked by reading, not run on hardware
//
#include <hip/hip_runtime.h>
#include <cstdint>
#include <cstddef>

// ---------------------------------------------------------------------------
// LoFTR encoder layer for gfx1250 (CDNA5, wave32).
// Heavy GEMMs: v_wmma_f32_16x16x32_bf16 (bf16 in, f32 accumulate), with the
// B (weight) tiles DMA'd into LDS by the Tensor Data Mover (double buffered,
// TENSORcnt synchronized) while waves convert/stage A tiles and compute.
// Fragment layout per CDNA5 ISA 7.12.2:
//   A (16x32, 16-bit): lane l: row = l%16, half h = l/16 holds
//       k in {8h..8h+7} (VGPR0-3) and {16+8h..16+8h+7} (VGPR4-7)
//   B (32x16, 16-bit): assumed symmetric (lane = column, same k pattern)
//   C/D (16x16 f32):   lane l: col = l%16, VGPR r: row = r + 8*(l/16)
// ---------------------------------------------------------------------------

typedef __bf16 bf16_t;
typedef __attribute__((ext_vector_type(8)))  float  v8f;
typedef __attribute__((ext_vector_type(8)))  bf16_t v8bf;
typedef __attribute__((ext_vector_type(16))) bf16_t v16bf;
typedef __attribute__((ext_vector_type(4)))  unsigned int u32x4;
typedef __attribute__((ext_vector_type(4)))  int i32x4;
typedef __attribute__((ext_vector_type(8)))  int i32x8;

#define NB   4
#define LL   4800
#define CC   256
#define TOK  (NB * LL)        // 19200
#define NHEAD 8
#define DH   32

static __device__ __forceinline__ bf16_t f2bf(float f) { return (bf16_t)f; }
static __device__ __forceinline__ float  bf2f(bf16_t b) { return (float)b; }

static __device__ __forceinline__ int kidx(int e, int half) {
  return (e < 8) ? (half * 8 + e) : (16 + half * 8 + (e - 8));
}

// Load one 16-lane-half fragment row: two ds/global b128 loads.
static __device__ __forceinline__ v16bf ld_frag16(const bf16_t* rowp, int half) {
  v8bf lo = *(const v8bf*)(rowp + half * 8);
  v8bf hi = *(const v8bf*)(rowp + 16 + half * 8);
  v16bf r;
#pragma unroll
  for (int i = 0; i < 8; ++i) { r[i] = lo[i]; r[i + 8] = hi[i]; }
  return r;
}

// ---------------------------------------------------------------------------
// TDM: DMA a [rows=256] x [k=32] bf16 tile (row stride = KD elements) from
// global memory into LDS at byte offset lds_off (contiguous 256x32 layout).
// Descriptor per CDNA5 ISA ch.8 (D# groups); 2D tensor, groups 2/3 zero.
// ---------------------------------------------------------------------------
static __device__ __forceinline__ void tdm_load_b_tile(const bf16_t* gptr,
                                                       unsigned lds_off,
                                                       int kd) {
  const unsigned long long ga = (unsigned long long)(uintptr_t)gptr;
  u32x4 g0;
  g0[0] = 1u;                                           // count=1, user mode
  g0[1] = lds_off;                                      // LDS byte address
  g0[2] = (unsigned)(ga & 0xFFFFFFFFu);                 // global_addr[31:0]
  g0[3] = (unsigned)((ga >> 32) & 0x01FFFFFFu) | (2u << 30);  // [56:32], type=2
  i32x8 g1;
  g1[0] = (int)(1u << 16);                              // data_size=1 (2 bytes)
  g1[1] = (int)(((unsigned)kd & 0xFFFFu) << 16);        // tensor_dim0[15:0]
  g1[2] = (int)((((unsigned)kd >> 16) & 0xFFFFu) | (256u << 16)); // d0 hi | tensor_dim1 lo
  g1[3] = (int)(32u << 16);                             // tensor_dim1 hi=0 | tile_dim0=32
  g1[4] = (int)256u;                                    // tile_dim1=256 | tile_dim2=0
  g1[5] = (int)kd;                                      // tensor_dim0_stride[31:0]
  g1[6] = 0;                                            // stride0 hi | stride1 lo
  g1[7] = 0;                                            // stride1 hi
  i32x4 z4 = {0, 0, 0, 0};
#if defined(__clang_major__) && (__clang_major__ >= 23)
  i32x8 z8 = {0, 0, 0, 0, 0, 0, 0, 0};
  __builtin_amdgcn_tensor_load_to_lds(g0, g1, z4, z4, z8, 0);
#else
  __builtin_amdgcn_tensor_load_to_lds(g0, g1, z4, z4, 0);
#endif
}

// ---------------------------------------------------------------------------
// Weight convert + transpose: in [K][N] f32 -> out [N][K] bf16
// ---------------------------------------------------------------------------
__global__ void k_wcvt(const float* __restrict__ in, bf16_t* __restrict__ out,
                       int K, int N) {
  int i = blockIdx.x * 256 + threadIdx.x;
  int k = i / N, n = i - k * N;
  out[(size_t)n * K + k] = f2bf(in[i]);
}

__global__ void k_zero(float* __restrict__ p, int n) {
  int i = blockIdx.x * 256 + threadIdx.x;
  if (i < n) p[i] = 0.f;
}

// ---------------------------------------------------------------------------
// Generic WMMA GEMM: C[M,NO] = A[M,KD] * B[KD,NO]
//   block = 256 threads (8 waves), tile = 128 rows x 256 cols, BK = 32
//   grid = (M/128, NOtotal/256)
//   B tiles arrive via TDM (double-buffered LDS, TENSORcnt sync by wave 0)
// ASRC: 0 = f32 A (lda=KD), 1 = bf16 A (lda=KD),
//       2 = concat: k<256 -> f32 Af (lda=256), k>=256 -> bf16 Ab (lda=256)
// EPI : 0 = f32 store, 1 = elu(x)+1 -> bf16, 2 = x*scale -> bf16, 3 = relu -> bf16
// ---------------------------------------------------------------------------
template <int KD, int ASRC, int EPI>
__global__ __launch_bounds__(256) void gemm_tile(
    const float* __restrict__ Af, const bf16_t* __restrict__ Ab,
    const bf16_t* __restrict__ Bt, float* __restrict__ outF,
    bf16_t* __restrict__ outB, int ldOut, float scaleEpi) {
  __shared__ bf16_t As[128 * 32];
  __shared__ bf16_t Bs[2][256 * 32];

  constexpr int KT = KD / 32;
  const int bm    = blockIdx.x * 128;
  const int nbase = blockIdx.y * 256;
  const int tid   = threadIdx.x;
  const int wid   = tid >> 5;
  const int lane  = tid & 31;
  const int cl    = lane & 15;
  const int half  = lane >> 4;

  v8f acc[16];
#pragma unroll
  for (int t = 0; t < 16; ++t)
#pragma unroll
    for (int r = 0; r < 8; ++r) acc[t][r] = 0.f;

  const int arow = tid >> 1;            // 0..127
  const int akb  = (tid & 1) * 16;      // 0 or 16
  const int gr   = bm + arow;

  const bf16_t* btile = Bt + (size_t)nbase * KD;
  const unsigned bs_off0 = (unsigned)(uintptr_t)(&Bs[0][0]);
  const unsigned bs_off1 = (unsigned)(uintptr_t)(&Bs[1][0]);

  // pre-issue TDM for the first B tile
  if (wid == 0) tdm_load_b_tile(btile, bs_off0, KD);

  for (int kt = 0; kt < KT; ++kt) {
    __syncthreads();  // previous iteration's LDS reads complete
    // ---- stage A tile (128 x 32) -> bf16 LDS ----
    {
      const int k0 = kt * 32 + akb;
      bf16_t* dst = As + arow * 32 + akb;
      if (ASRC == 0 || (ASRC == 2 && k0 < 256)) {
        const int lda = (ASRC == 2) ? 256 : KD;
        const float4* s = (const float4*)(Af + (size_t)gr * lda + k0);
        float tmp[16];
        *(float4*)(tmp + 0)  = s[0];
        *(float4*)(tmp + 4)  = s[1];
        *(float4*)(tmp + 8)  = s[2];
        *(float4*)(tmp + 12) = s[3];
        v8bf o0, o1;
#pragma unroll
        for (int i = 0; i < 8; ++i) { o0[i] = f2bf(tmp[i]); o1[i] = f2bf(tmp[8 + i]); }
        *(v8bf*)(dst + 0) = o0;
        *(v8bf*)(dst + 8) = o1;
        if (ASRC == 0 && kt + 1 < KT)
          __builtin_prefetch(Af + (size_t)gr * lda + k0 + 32, 0, 1);
      } else {
        const int lda = (ASRC == 2) ? 256 : KD;
        const int kk  = (ASRC == 2) ? (k0 - 256) : k0;
        const uint4* s = (const uint4*)(Ab + (size_t)gr * lda + kk);
        uint4* d = (uint4*)dst;
        d[0] = s[0];
        d[1] = s[1];
      }
    }
    // ---- TDM: prefetch next B tile, wait for current one ----
    if (wid == 0) {
      if (kt + 1 < KT) {
        tdm_load_b_tile(btile + (kt + 1) * 32,
                        ((kt + 1) & 1) ? bs_off1 : bs_off0, KD);
        __builtin_amdgcn_s_wait_tensorcnt(1);  // current tile landed
      } else {
        __builtin_amdgcn_s_wait_tensorcnt(0);
      }
    }
    __syncthreads();  // As staged + Bs[kt&1] visible to all waves
    // ---- compute: wave w = 16 rows x 256 cols (16 WMMA tiles) ----
    const bf16_t* bbuf = Bs[kt & 1];
    const bf16_t* ap = As + (wid * 16 + cl) * 32;
    v16bf a = ld_frag16(ap, half);
#pragma unroll
    for (int tt = 0; tt < 16; ++tt) {
      const bf16_t* bp = bbuf + (tt * 16 + cl) * 32;
      v16bf b = ld_frag16(bp, half);
      acc[tt] = __builtin_amdgcn_wmma_f32_16x16x32_bf16(
          false, a, false, b, (short)0, acc[tt], false, false);
    }
  }

  // ---- epilogue ----
#pragma unroll
  for (int tt = 0; tt < 16; ++tt) {
    const int col = nbase + tt * 16 + cl;
#pragma unroll
    for (int r = 0; r < 8; ++r) {
      const int grow = bm + wid * 16 + r + 8 * half;
      float v = acc[tt][r];
      if (EPI == 0) {
        outF[(size_t)grow * ldOut + col] = v;
      } else {
        float o;
        if (EPI == 1)      o = (v > 0.f) ? (v + 1.f) : __expf(v);  // elu(x)+1
        else if (EPI == 2) o = v * scaleEpi;
        else               o = (v > 0.f) ? v : 0.f;                // relu
        outB[(size_t)grow * ldOut + col] = f2bf(o);
      }
    }
  }
}

// ---------------------------------------------------------------------------
// KV[n,h,d,v] = sum_s K[n,s,h,d] * vals[n,s,h,v];  Ksum[n,h,d] = sum_s K
// grid = (N*H, 30), block = 256. Each block reduces 160 s-rows, atomics to f32.
// ---------------------------------------------------------------------------
__global__ __launch_bounds__(256) void k_kv(const bf16_t* __restrict__ Kb,
                                            const bf16_t* __restrict__ Vb,
                                            float* __restrict__ KV,
                                            float* __restrict__ Ksum) {
  __shared__ bf16_t Kl[32 * 32];
  __shared__ bf16_t Vl[32 * 32];
  const int g = blockIdx.x, n = g >> 3, h = g & 7;
  const int s0 = blockIdx.y * 160;
  const int tid = threadIdx.x;
  const int d  = tid >> 3;
  const int vg = tid & 7;
  const int v0 = vg * 4;
  const int ls = tid >> 3, ldb = (tid & 7) * 4;  // LDS-fill coords

  float acc[4] = {0.f, 0.f, 0.f, 0.f};
  float ks = 0.f;

  for (int t = 0; t < 5; ++t) {
    const int sb = s0 + t * 32;
    __syncthreads();
    {
      const size_t base = ((size_t)(n * LL + sb + ls)) * CC + h * DH + ldb;
      *(unsigned long long*)(Kl + ls * 32 + ldb) = *(const unsigned long long*)(Kb + base);
      *(unsigned long long*)(Vl + ls * 32 + ldb) = *(const unsigned long long*)(Vb + base);
    }
    __syncthreads();
#pragma unroll 8
    for (int s = 0; s < 32; ++s) {
      const float kd = bf2f(Kl[s * 32 + d]);
      ks += kd;
#pragma unroll
      for (int j = 0; j < 4; ++j) acc[j] += kd * bf2f(Vl[s * 32 + v0 + j]);
    }
  }
  float* kvp = KV + ((size_t)(n * NHEAD + h) * DH + d) * DH + v0;
#pragma unroll
  for (int j = 0; j < 4; ++j) atomicAdd(kvp + j, acc[j]);
  if (vg == 0) atomicAdd(Ksum + (n * NHEAD + h) * DH + d, ks);
}

// ---------------------------------------------------------------------------
// message[l, h*32+v] = (Q[l,h,:] @ KV[h]) * S / (Q[l,h,:].Ksum[h] + eps)
// grid = (1200 token-tiles, 8 heads), block = 32 (one wave, 2 WMMAs)
// ---------------------------------------------------------------------------
__global__ __launch_bounds__(32) void k_attn(const bf16_t* __restrict__ Qb,
                                             const float* __restrict__ KV,
                                             const float* __restrict__ Ksum,
                                             bf16_t* __restrict__ msg) {
  __shared__ float zb[16];
  const int tile = blockIdx.x;          // 0..1199
  const int h    = blockIdx.y;          // 0..7
  const int lane = threadIdx.x;
  const int n    = tile / (LL / 16);
  const int row0 = tile * 16;
  const int cl   = lane & 15;
  const int half = lane >> 4;

  // A fragment: 16 tokens x 32 head-dims (contiguous per row)
  const bf16_t* qp = Qb + (size_t)(row0 + cl) * CC + h * DH;
  v16bf a = ld_frag16(qp, half);

  // Z = S / (Q . Ksum + eps), per token row
  const float* ks = Ksum + (n * NHEAD + h) * DH;
  float part = 0.f;
#pragma unroll
  for (int e = 0; e < 16; ++e) part += bf2f(a[e]) * ks[kidx(e, half)];
  part += __shfl_xor(part, 16, 32);
  if (lane < 16) zb[lane] = (float)LL / (part + 1e-6f);
  __syncthreads();

  // B fragments from f32 KV (L2 hot), converted to bf16
  const float* kv = KV + (size_t)(n * NHEAD + h) * (DH * DH);
  v16bf b0, b1;
#pragma unroll
  for (int e = 0; e < 16; ++e) {
    const int k = kidx(e, half);
    b0[e] = f2bf(kv[k * DH + cl]);
    b1[e] = f2bf(kv[k * DH + 16 + cl]);
  }
  v8f acc0, acc1;
#pragma unroll
  for (int r = 0; r < 8; ++r) { acc0[r] = 0.f; acc1[r] = 0.f; }
  acc0 = __builtin_amdgcn_wmma_f32_16x16x32_bf16(false, a, false, b0, (short)0,
                                                 acc0, false, false);
  acc1 = __builtin_amdgcn_wmma_f32_16x16x32_bf16(false, a, false, b1, (short)0,
                                                 acc1, false, false);
#pragma unroll
  for (int r = 0; r < 8; ++r) {
    const int m = r + 8 * half;
    const float z = zb[m];
    const size_t rb = (size_t)(row0 + m) * CC + h * DH;
    msg[rb + cl]      = f2bf(acc0[r] * z);
    msg[rb + 16 + cl] = f2bf(acc1[r] * z);
  }
}

// ---------------------------------------------------------------------------
// LayerNorm (one wave per 256-wide row), bf16 output
// ---------------------------------------------------------------------------
__global__ __launch_bounds__(256) void k_layernorm(const float* __restrict__ in,
                                                   const float* __restrict__ g,
                                                   const float* __restrict__ b,
                                                   bf16_t* __restrict__ out) {
  const int row  = blockIdx.x * 8 + (threadIdx.x >> 5);
  const int lane = threadIdx.x & 31;
  const float* p = in + (size_t)row * CC + lane * 8;
  float v[8];
  *(float4*)(v + 0) = ((const float4*)p)[0];
  *(float4*)(v + 4) = ((const float4*)p)[1];
  float s = 0.f;
#pragma unroll
  for (int i = 0; i < 8; ++i) s += v[i];
#pragma unroll
  for (int off = 16; off > 0; off >>= 1) s += __shfl_xor(s, off, 32);
  const float mu = s * (1.f / 256.f);
  float q = 0.f;
#pragma unroll
  for (int i = 0; i < 8; ++i) { const float d = v[i] - mu; q += d * d; }
#pragma unroll
  for (int off = 16; off > 0; off >>= 1) q += __shfl_xor(q, off, 32);
  const float rs = rsqrtf(q * (1.f / 256.f) + 1e-5f);
#pragma unroll
  for (int i = 0; i < 8; ++i) {
    const int c = lane * 8 + i;
    out[(size_t)row * CC + c] = f2bf((v[i] - mu) * rs * g[c] + b[c]);
  }
}

// LayerNorm + residual, f32 output (final)
__global__ __launch_bounds__(256) void k_layernorm_res(
    const float* __restrict__ in, const float* __restrict__ g,
    const float* __restrict__ b, const float* __restrict__ xin,
    float* __restrict__ out) {
  const int row  = blockIdx.x * 8 + (threadIdx.x >> 5);
  const int lane = threadIdx.x & 31;
  const float* p = in + (size_t)row * CC + lane * 8;
  float v[8];
  *(float4*)(v + 0) = ((const float4*)p)[0];
  *(float4*)(v + 4) = ((const float4*)p)[1];
  float s = 0.f;
#pragma unroll
  for (int i = 0; i < 8; ++i) s += v[i];
#pragma unroll
  for (int off = 16; off > 0; off >>= 1) s += __shfl_xor(s, off, 32);
  const float mu = s * (1.f / 256.f);
  float q = 0.f;
#pragma unroll
  for (int i = 0; i < 8; ++i) { const float d = v[i] - mu; q += d * d; }
#pragma unroll
  for (int off = 16; off > 0; off >>= 1) q += __shfl_xor(q, off, 32);
  const float rs = rsqrtf(q * (1.f / 256.f) + 1e-5f);
#pragma unroll
  for (int i = 0; i < 8; ++i) {
    const int c = lane * 8 + i;
    const size_t idx = (size_t)row * CC + c;
    out[idx] = xin[idx] + (v[i] - mu) * rs * g[c] + b[c];
  }
}

// ---------------------------------------------------------------------------
// Host launcher
// ---------------------------------------------------------------------------
extern "C" void kernel_launch(void* const* d_in, const int* in_sizes, int n_in,
                              void* d_out, int out_size, void* d_ws,
                              size_t ws_size, hipStream_t stream) {
  (void)in_sizes; (void)n_in; (void)out_size; (void)ws_size;
  const float* x   = (const float*)d_in[0];
  const float* src = (const float*)d_in[1];
  const float* Wq  = (const float*)d_in[2];
  const float* Wk  = (const float*)d_in[3];
  const float* Wv  = (const float*)d_in[4];
  const float* Wm  = (const float*)d_in[5];
  const float* W1  = (const float*)d_in[6];
  const float* W2  = (const float*)d_in[7];
  const float* g1  = (const float*)d_in[8];
  const float* b1  = (const float*)d_in[9];
  const float* g2  = (const float*)d_in[10];
  const float* b2  = (const float*)d_in[11];
  float* out = (float*)d_out;

  // workspace carve (bytes, all 128B aligned)
  char* w = (char*)d_ws;
  const size_t SZ_TOKBF = (size_t)TOK * CC * 2;  // 9,830,400
  bf16_t* Qb   = (bf16_t*)(w + 0);
  bf16_t* Kb   = (bf16_t*)(w + SZ_TOKBF);
  bf16_t* Vb   = (bf16_t*)(w + 2 * SZ_TOKBF);
  bf16_t* msg  = (bf16_t*)(w + 3 * SZ_TOKBF);
  bf16_t* msgn = (bf16_t*)(w + 4 * SZ_TOKBF);
  size_t off = 5 * SZ_TOKBF;
  bf16_t* Wqt = (bf16_t*)(w + off); off += 256 * 256 * 2;
  bf16_t* Wkt = (bf16_t*)(w + off); off += 256 * 256 * 2;
  bf16_t* Wvt = (bf16_t*)(w + off); off += 256 * 256 * 2;
  bf16_t* Wmt = (bf16_t*)(w + off); off += 256 * 256 * 2;
  bf16_t* W1t = (bf16_t*)(w + off); off += 512 * 512 * 2;
  bf16_t* W2t = (bf16_t*)(w + off); off += 256 * 512 * 2;
  float*  KVf = (float*)(w + off);  off += NB * NHEAD * DH * DH * 4;
  float*  Ksm = (float*)(w + off);  off += NB * NHEAD * DH * 4;
  float*  mm  = (float*)(w + off);  off += (size_t)TOK * CC * 4;  // also reused as h2
  bf16_t* h1  = Kb;  // aliases Kb+Vb (dead after k_kv), 19200*512 bf16

  const dim3 B256(256);

  // 1) weight prep
  k_wcvt<<<256,  B256, 0, stream>>>(Wq, Wqt, 256, 256);
  k_wcvt<<<256,  B256, 0, stream>>>(Wk, Wkt, 256, 256);
  k_wcvt<<<256,  B256, 0, stream>>>(Wv, Wvt, 256, 256);
  k_wcvt<<<256,  B256, 0, stream>>>(Wm, Wmt, 256, 256);
  k_wcvt<<<1024, B256, 0, stream>>>(W1, W1t, 512, 512);
  k_wcvt<<<512,  B256, 0, stream>>>(W2, W2t, 512, 256);
  k_zero<<<(NB * NHEAD * DH * (DH + 1) + 255) / 256, B256, 0, stream>>>(
      KVf, NB * NHEAD * DH * (DH + 1));

  // 2) Q = elu(x@Wq)+1, K = elu(src@Wk)+1, V = (src@Wv)/S   (bf16)
  gemm_tile<256, 0, 1><<<dim3(TOK / 128, 1), B256, 0, stream>>>(
      x, nullptr, Wqt, nullptr, Qb, CC, 0.f);
  gemm_tile<256, 0, 1><<<dim3(TOK / 128, 1), B256, 0, stream>>>(
      src, nullptr, Wkt, nullptr, Kb, CC, 0.f);
  gemm_tile<256, 0, 2><<<dim3(TOK / 128, 1), B256, 0, stream>>>(
      src, nullptr, Wvt, nullptr, Vb, CC, 1.0f / (float)LL);

  // 3) KV + Ksum reduction
  k_kv<<<dim3(NB * NHEAD, 30), B256, 0, stream>>>(Kb, Vb, KVf, Ksm);

  // 4) attention message (bf16)
  k_attn<<<dim3(TOK / 16, NHEAD), dim3(32), 0, stream>>>(Qb, KVf, Ksm, msg);

  // 5) merge: msg @ Wm -> f32, then LN1 -> bf16
  gemm_tile<256, 1, 0><<<dim3(TOK / 128, 1), B256, 0, stream>>>(
      nullptr, msg, Wmt, mm, nullptr, CC, 0.f);
  k_layernorm<<<TOK / 8, B256, 0, stream>>>(mm, g1, b1, msgn);

  // 6) MLP: relu(concat(x,msgn) @ W1) -> bf16 h1; h1 @ W2 -> f32; LN2 + residual
  gemm_tile<512, 2, 3><<<dim3(TOK / 128, 2), B256, 0, stream>>>(
      x, msgn, W1t, nullptr, h1, 512, 0.f);
  gemm_tile<512, 1, 0><<<dim3(TOK / 128, 1), B256, 0, stream>>>(
      nullptr, h1, W2t, mm, nullptr, CC, 0.f);
  k_layernorm_res<<<TOK / 8, B256, 0, stream>>>(mm, g2, b2, x, out);
}
